// Canny_76862734729366
// MI455X (gfx1250) — matrix operand order repaired and checked by reading, not compiled, not run on hardware
//
#include <hip/hip_runtime.h>
#include <math.h>

// ---------------------------------------------------------------- constants
#define IMG_H 1024
#define IMG_W 1024
#define NBATCH 4
#define HW (IMG_H * IMG_W)
#define CANNY_EPS 1e-6f
#define HYST_PAIRS 2   // 4 ping-pong steps

typedef __attribute__((ext_vector_type(16))) _Float16 v16h;
typedef __attribute__((ext_vector_type(8)))  float    v8f;

// gaussian(ks=5, sigma=1), L1-normalized -- compile-time immediates
constexpr float GWc[5] = {0.05448868454511f, 0.24420134200323f,
                          0.40261994689332f, 0.24420134200323f,
                          0.05448868454511f};
// canny NMS neighbor offsets packed as 2-bit (d+1) fields: dy={0,1,1,1,0,-1,-1,-1}, dx={1,1,0,-1,-1,-1,0,1}
#define PACK_DY 425
#define PACK_DX 36890

__device__ __forceinline__ int reflect_idx(int p, int n) {
  if (p < 0) p = -p;                 // jnp 'reflect' (no edge dup), halo<=2<n
  if (p >= n) p = 2 * n - 2 - p;
  return p;
}
__device__ __forceinline__ int clamp_idx(int p, int n) {
  return min(max(p, 0), n - 1);
}

// ---------------------------------------------------------------- stage 0+1: global max + grayscale (fused)
__global__ void init_max_kernel(float* m) {
  *(unsigned int*)m = 0u;            // input uniform[0,1): non-negative
}

__global__ __launch_bounds__(256) void gray_max_kernel(const float* __restrict__ x,
                                                       float* __restrict__ gray,
                                                       float* __restrict__ m) {
  int i = blockIdx.x * 256 + threadIdx.x;        // exactly NBATCH*HW threads
  int b = i >> 20;                               // HW = 1<<20
  int p = i & (HW - 1);
  const float* px = x + (size_t)b * 3 * HW + p;
  float r = px[0], g = px[HW], bb = px[2 * HW];
  gray[i] = 0.299f * r + 0.587f * g + 0.114f * bb;
  float v = fmaxf(r, fmaxf(g, bb));              // max over all channels == global max

  __shared__ float s[256];
  s[threadIdx.x] = v;
  __syncthreads();
  for (int o = 128; o > 0; o >>= 1) {
    if (threadIdx.x < (unsigned)o) s[threadIdx.x] = fmaxf(s[threadIdx.x], s[threadIdx.x + o]);
    __syncthreads();
  }
  if (threadIdx.x == 0)
    atomicMax((unsigned int*)m, __float_as_uint(s[0]));   // bit order == float order for >=0
}

// ---------------------------------------------------------------- stage 2: 5x5 gaussian via WMMA
// 4 waves per block, each wave one 16x16 quadrant of a 32x32 tile:
//   D(16x16) = sum_{j=0..4}  A_j(16x32) x B(32x16)
//   A_j[r][k] = GW[j] * gray_reflect(r0+qy+r+j-2, c0+qx+k-2)
//   B[k][n]   = GW[k-n] for 0<=k-n<=4, else 0 (zero rows k>=20 nullify A's tail)
// LDS tile padded to 48 cols; cols 36..47 zeroed so all A reads are finite (WMMA propagates NaN).
__global__ __launch_bounds__(128) void blur_wmma_kernel(const float* __restrict__ gray,
                                                        float* __restrict__ blur) {
  __shared__ float g[36][48];
  const int tid = threadIdx.x;
  const int lane = tid & 31;
  const int wv = tid >> 5;
  const int qx = (wv & 1) * 16, qy = (wv >> 1) * 16;
  const int c0 = blockIdx.x * 32, r0 = blockIdx.y * 32, b = blockIdx.z;
  const float* gp = gray + (size_t)b * HW;

  for (int idx = tid; idx < 36 * 36; idx += 128) {
    int ry = idx / 36, rx = idx % 36;
    g[ry][rx] = gp[(size_t)reflect_idx(r0 + ry - 2, IMG_H) * IMG_W +
                   reflect_idx(c0 + rx - 2, IMG_W)];
  }
  for (int idx = tid; idx < 36 * 12; idx += 128) {
    g[idx / 12][36 + idx % 12] = 0.f;
  }
  __syncthreads();

  // B fragment: lane%16 = N; comps 0..7 -> K=8*hi.., comps 8..15 -> K=16+8*hi..; weights via cndmask chain
  const int n = lane & 15;
  const int hi = lane >> 4;
  v16h bf;
#pragma unroll
  for (int c = 0; c < 16; ++c) {
    int d = (8 * hi + c + (c >= 8 ? 8 : 0)) - n;
    float w = 0.f;
#pragma unroll
    for (int t = 0; t < 5; ++t) w = (d == t) ? GWc[t] : w;
    bf[c] = (_Float16)w;
  }

  // A fragments: lane%16 = M; comps 0..7 -> K=8*hi.., comps 8..15 -> K=16+8*hi..
  v8f acc = {};
#pragma unroll
  for (int j = 0; j < 5; ++j) {
    const float wj = GWc[j];
    const float* row = &g[qy + n + j][qx + 8 * hi];   // n == lane&15 == M
    float4 p0 = *(const float4*)(row);
    float4 p1 = *(const float4*)(row + 4);
    float4 p2 = *(const float4*)(row + 16);
    float4 p3 = *(const float4*)(row + 20);
    v16h af;
    af[0]  = (_Float16)(wj * p0.x); af[1]  = (_Float16)(wj * p0.y);
    af[2]  = (_Float16)(wj * p0.z); af[3]  = (_Float16)(wj * p0.w);
    af[4]  = (_Float16)(wj * p1.x); af[5]  = (_Float16)(wj * p1.y);
    af[6]  = (_Float16)(wj * p1.z); af[7]  = (_Float16)(wj * p1.w);
    af[8]  = (_Float16)(wj * p2.x); af[9]  = (_Float16)(wj * p2.y);
    af[10] = (_Float16)(wj * p2.z); af[11] = (_Float16)(wj * p2.w);
    af[12] = (_Float16)(wj * p3.x); af[13] = (_Float16)(wj * p3.y);
    af[14] = (_Float16)(wj * p3.z); af[15] = (_Float16)(wj * p3.w);
    acc = __builtin_amdgcn_wmma_f32_16x16x32_f16(false, af, false, bf,
                                                 (short)0, acc, false, false);
  }

  // D: VGPR i -> M = i + 8*hi2 where lanes 16-31 carry M+8, N = lane&15
  float* op = blur + (size_t)b * HW;
  const int mbase = hi * 8;
#pragma unroll
  for (int i = 0; i < 8; ++i)
    op[(size_t)(r0 + qy + mbase + i) * IMG_W + (c0 + qx + n)] = acc[i];
}

// ---------------------------------------------------------------- stage 3: sobel + NMS + thresholds (32x32 tile)
__global__ __launch_bounds__(256) void sobel_nms_kernel(const float* __restrict__ blur,
                                                        const float* __restrict__ maxv,
                                                        float* __restrict__ edges) {
  __shared__ float bl[36][36];   // blurred, replicate-clamped, halo 2
  __shared__ float mg[34][34];   // magnitude, halo 1 (0 outside image = zero-pad shift)
  const int tid = threadIdx.x;
  const int tx = tid & 15, ty = tid >> 4;
  const int c0 = blockIdx.x * 32, r0 = blockIdx.y * 32, b = blockIdx.z;
  const float* bp = blur + (size_t)b * HW;

  for (int idx = tid; idx < 36 * 36; idx += 256) {
    int ry = idx / 36, rx = idx % 36;
    bl[ry][rx] = bp[(size_t)clamp_idx(r0 + ry - 2, IMG_H) * IMG_W +
                    clamp_idx(c0 + rx - 2, IMG_W)];
  }
  __syncthreads();

  for (int idx = tid; idx < 34 * 34; idx += 256) {
    int my = idx / 34, mx = idx % 34;
    int Y = r0 + my - 1, X = c0 + mx - 1;
    float mval = 0.f;
    if (Y >= 0 && Y < IMG_H && X >= 0 && X < IMG_W) {
      int ry = my + 1, rx = mx + 1;
      float a00 = bl[ry - 1][rx - 1], a01 = bl[ry - 1][rx], a02 = bl[ry - 1][rx + 1];
      float a10 = bl[ry][rx - 1],                           a12 = bl[ry][rx + 1];
      float a20 = bl[ry + 1][rx - 1], a21 = bl[ry + 1][rx], a22 = bl[ry + 1][rx + 1];
      float gx = (a02 + 2.f * a12 + a22 - a00 - 2.f * a10 - a20) * 0.125f;
      float gy = (a20 + 2.f * a21 + a22 - a00 - 2.f * a01 - a02) * 0.125f;
      mval = sqrtf(gx * gx + gy * gy + CANNY_EPS);
    }
    mg[my][mx] = mval;
  }
  __syncthreads();

  float mv = maxv[0];
  float low_t = 0.1f * mv, high_t = 0.4f * mv;

#pragma unroll
  for (int sy = 0; sy < 2; ++sy)
#pragma unroll
    for (int sx = 0; sx < 2; ++sx) {
      const int py = ty + 16 * sy, px = tx + 16 * sx;
      const int ry = py + 2, rx = px + 2;
      float a00 = bl[ry - 1][rx - 1], a01 = bl[ry - 1][rx], a02 = bl[ry - 1][rx + 1];
      float a10 = bl[ry][rx - 1],                           a12 = bl[ry][rx + 1];
      float a20 = bl[ry + 1][rx - 1], a21 = bl[ry + 1][rx], a22 = bl[ry + 1][rx + 1];
      float gx = (a02 + 2.f * a12 + a22 - a00 - 2.f * a10 - a20) * 0.125f;
      float gy = (a20 + 2.f * a21 + a22 - a00 - 2.f * a01 - a02) * 0.125f;
      float mc = mg[py + 1][px + 1];

      // quantized direction in [-4,4]; rintf == round-half-even (jnp.round)
      float ang = rintf(atan2f(gy, gx) * 1.2732395447351628f);   // (180/pi)/45 = 4/pi
      int ai = (int)ang;
      int ip = ((ai % 8) + 8) % 8;
      int in_ = (((ai + 4) % 8) + 8) % 8;
      int dyp = ((PACK_DY >> (2 * ip)) & 3) - 1, dxp = ((PACK_DX >> (2 * ip)) & 3) - 1;
      int dyn = ((PACK_DY >> (2 * in_)) & 3) - 1, dxn = ((PACK_DX >> (2 * in_)) & 3) - 1;

      float cp = mc - mg[py + 1 + dyp][px + 1 + dxp];
      float cn = mc - mg[py + 1 + dyn][px + 1 + dxn];
      float m2 = (fminf(cp, cn) > 0.f) ? mc : 0.f;

      float lo  = (m2 > low_t)  ? m2 : 0.f;    // F.threshold: strict >
      float hiv = (m2 > high_t) ? m2 : 0.f;
      edges[(size_t)b * HW + (size_t)(r0 + py) * IMG_W + (c0 + px)] = 0.5f * lo + 0.5f * hiv;
    }
}

// ---------------------------------------------------------------- stage 4: hysteresis step (32x32 tile)
__global__ __launch_bounds__(256) void hyst_kernel(const float* __restrict__ ein,
                                                   float* __restrict__ eout,
                                                   int write_hm) {
  __shared__ float t[34][34];
  const int tid = threadIdx.x;
  const int tx = tid & 15, ty = tid >> 4;
  const int c0 = blockIdx.x * 32, r0 = blockIdx.y * 32, b = blockIdx.z;
  const float* ep = ein + (size_t)b * HW;

  for (int idx = tid; idx < 34 * 34; idx += 256) {
    int my = idx / 34, mx = idx % 34;
    int Y = r0 + my - 1, X = c0 + mx - 1;
    t[my][mx] = (Y >= 0 && Y < IMG_H && X >= 0 && X < IMG_W)
                    ? ep[(size_t)Y * IMG_W + X]
                    : -INFINITY;   // reduce_window init = -inf
  }
  __syncthreads();

#pragma unroll
  for (int sy = 0; sy < 2; ++sy)
#pragma unroll
    for (int sx = 0; sx < 2; ++sx) {
      const int py = ty + 16 * sy, px = tx + 16 * sx;
      float e = t[py + 1][px + 1];
      float nmax = -INFINITY;
#pragma unroll
      for (int dy = 0; dy < 3; ++dy)
#pragma unroll
        for (int dx = 0; dx < 3; ++dx)
          nmax = fmaxf(nmax, t[py + dy][px + dx]);

      float weak   = (e == 0.5f) ? 1.f : 0.f;
      float strong = (e == 1.0f) ? 1.f : 0.f;
      float hm = ((nmax == 1.0f) ? 1.f : 0.f) * weak + strong;
      float ne = hm + ((hm == 0.f) ? 1.f : 0.f) * weak * 0.5f;
      eout[(size_t)b * HW + (size_t)(r0 + py) * IMG_W + (c0 + px)] = write_hm ? hm : ne;
    }
}

// ---------------------------------------------------------------- launch
extern "C" void kernel_launch(void* const* d_in, const int* in_sizes, int n_in,
                              void* d_out, int out_size, void* d_ws, size_t ws_size,
                              hipStream_t stream) {
  (void)in_sizes; (void)n_in; (void)out_size; (void)ws_size;
  const float* x = (const float*)d_in[0];          // (4,3,1024,1024) f32
  float* out = (float*)d_out;                      // (4,1,1024,1024) f32

  char* ws = (char*)d_ws;
  float* maxv = (float*)ws;                        // 1 float (256B slot)
  float* bufA = (float*)(ws + 256);                // 16 MiB: gray -> edges ping
  float* bufB = bufA + (size_t)NBATCH * HW;        // 16 MiB: blurred -> edges pong

  // 0+1) global max + grayscale, fused single pass over input
  init_max_kernel<<<1, 1, 0, stream>>>(maxv);
  gray_max_kernel<<<(NBATCH * HW) / 256, 256, 0, stream>>>(x, bufA, maxv);

  // 2) gaussian blur on the matrix pipe (v_wmma_f32_16x16x32_f16), 4 waves / 32x32 tile
  dim3 tiles32(IMG_W / 32, IMG_H / 32, NBATCH);
  blur_wmma_kernel<<<tiles32, 128, 0, stream>>>(bufA, bufB);

  // 3) sobel + magnitude/angle + NMS + double threshold
  sobel_nms_kernel<<<tiles32, 256, 0, stream>>>(bufB, maxv, bufA);

  // 4) hysteresis: fixed even ping-pong (L2-resident), then emit hm
  for (int i = 0; i < HYST_PAIRS; ++i) {
    hyst_kernel<<<tiles32, 256, 0, stream>>>(bufA, bufB, 0);
    hyst_kernel<<<tiles32, 256, 0, stream>>>(bufB, bufA, 0);
  }
  hyst_kernel<<<tiles32, 256, 0, stream>>>(bufA, out, 1);
}
